// MultiHeadDotProductAttentionQKV_15710990368815
// MI455X (gfx1250) — compile-verified, matrix operand order checked by reading, not run on hardware
//
#include <hip/hip_runtime.h>

// ---------------------------------------------------------------------------
// MI455X (gfx1250) multi-head attention, bf16 WMMA pipeline with async
// (DMA) global->LDS tile copies and double-buffered GEMM main loops.
// B=4, S=2048, D=1024, H=16, HD=64.
// ---------------------------------------------------------------------------

typedef __attribute__((ext_vector_type(16))) __bf16 v16bf;
typedef __attribute__((ext_vector_type(8)))  __bf16 v8bf;
typedef __attribute__((ext_vector_type(8)))  float  v8f;

#define B_   4
#define S_   2048
#define D_   1024
#define H_   16
#define HD_  64
#define M_   (B_ * S_)        // 8192
#define NQKV (3 * H_ * HD_)   // 3072

// Build a 16-element bf16 fragment from two contiguous 8-element (16-byte) chunks.
__device__ __forceinline__ v16bf frag16(const __bf16* p0, const __bf16* p1) {
    v8bf lo = *(const v8bf*)p0;
    v8bf hi = *(const v8bf*)p1;
    return __builtin_shufflevector(lo, hi, 0, 1, 2, 3, 4, 5, 6, 7,
                                           8, 9, 10, 11, 12, 13, 14, 15);
}

__device__ __forceinline__ v8f wmma_bf16(v16bf a, v16bf b, v8f c) {
    // D = A(16x32) * B(32x16) + C, fp32 accumulate.
    return __builtin_amdgcn_wmma_f32_16x16x32_bf16(false, a, false, b,
                                                   (short)0, c, false, false);
}

// gfx1250 async DMA: 16B global -> LDS, tracked by ASYNCcnt. The LDS address
// VGPR holds the offset within the wave's LDS allocation, which is the low
// 32 bits of the flat shared-aperture pointer.
__device__ __forceinline__ void async_copy_b128(void* lds_dst, const void* gsrc) {
    uint32_t lds_off = (uint32_t)(uintptr_t)lds_dst;
    asm volatile("global_load_async_to_lds_b128 %0, %1, off"
                 :: "v"(lds_off), "v"((uint64_t)(uintptr_t)gsrc)
                 : "memory");
}
__device__ __forceinline__ void wait_async0() {
    asm volatile("s_wait_asynccnt 0" ::: "memory");
}

// ---------------------------------------------------------------------------
// Conversion kernels (fp32 -> bf16), plain and transposing.
// ---------------------------------------------------------------------------
__global__ void cvt_f32_bf16(const float* __restrict__ in,
                             __bf16* __restrict__ out, int n) {
    int i = blockIdx.x * blockDim.x + threadIdx.x;
    if (i < n) out[i] = (__bf16)in[i];
}

// in: [rows][cols] fp32 -> out: [cols][rows] bf16
__global__ void cvt_transpose_bf16(const float* __restrict__ in,
                                   __bf16* __restrict__ out,
                                   int rows, int cols) {
    int i = blockIdx.x * blockDim.x + threadIdx.x;
    int n = rows * cols;
    if (i < n) {
        int r = i / cols;
        int c = i - r * cols;
        out[(size_t)c * rows + r] = (__bf16)in[i];
    }
}

// ---------------------------------------------------------------------------
// Shared GEMM core: block tile 128(M) x 64(N), K-step 64, double-buffered LDS
// filled by async DMA copies. 8 waves; each wave computes 16x64 (4 accums).
// A  [.][D_] bf16 row-major, Bt [.][D_] bf16 (B pre-transposed: N rows, K
// contiguous) -> both tiles use identical fragment addressing.
// ---------------------------------------------------------------------------
#define GEMM_CORE(A_, Bt_, ACC_)                                              \
    __shared__ __bf16 As[2][128][72];   /* pad 8 -> 144B stride (16B mult) */ \
    __shared__ __bf16 Bs[2][64][72];                                          \
    const int t    = threadIdx.x;                                             \
    const int wv   = t >> 5;                                                  \
    const int lane = t & 31;                                                  \
    const int lr   = lane & 15;                                               \
    const int hs   = lane >> 4;                                               \
    const int mBase = blockIdx.y * 128;                                       \
    const int nBase = blockIdx.x * 64;                                        \
    v8f ACC_[4];                                                              \
    _Pragma("unroll")                                                         \
    for (int i = 0; i < 4; ++i)                                               \
        _Pragma("unroll")                                                     \
        for (int r = 0; r < 8; ++r) ACC_[i][r] = 0.f;                         \
    auto issue_tile = [&](int k0, int buf) {                                  \
        _Pragma("unroll")                                                     \
        for (int i = 0; i < 4; ++i) {  /* A: 128x64 = 1024 16B chunks */      \
            int c = t + i * 256;                                              \
            int row = c >> 3, col = (c & 7) * 8;                              \
            async_copy_b128(&As[buf][row][col],                               \
                            &A_[(size_t)(mBase + row) * D_ + k0 + col]);      \
        }                                                                     \
        _Pragma("unroll")                                                     \
        for (int i = 0; i < 2; ++i) {  /* B: 64x64 = 512 16B chunks */        \
            int c = t + i * 256;                                              \
            int row = c >> 3, col = (c & 7) * 8;                              \
            async_copy_b128(&Bs[buf][row][col],                               \
                            &Bt_[(size_t)(nBase + row) * D_ + k0 + col]);     \
        }                                                                     \
    };                                                                        \
    issue_tile(0, 0);                                                         \
    for (int it = 0; it < D_ / 64; ++it) {                                    \
        wait_async0();          /* this wave's tile `it` landed in LDS   */   \
        __syncthreads();        /* everyone's landed; prev reads retired */   \
        if (it + 1 < D_ / 64) issue_tile((it + 1) * 64, (it + 1) & 1);        \
        const int buf = it & 1;                                               \
        _Pragma("unroll")                                                     \
        for (int kk = 0; kk < 64; kk += 32) {                                 \
            v16bf a  = frag16(&As[buf][wv * 16 + lr][kk + hs * 8],            \
                              &As[buf][wv * 16 + lr][kk + 16 + hs * 8]);      \
            v16bf b0 = frag16(&Bs[buf][lr][kk + hs * 8],                      \
                              &Bs[buf][lr][kk + 16 + hs * 8]);                \
            v16bf b1 = frag16(&Bs[buf][16 + lr][kk + hs * 8],                 \
                              &Bs[buf][16 + lr][kk + 16 + hs * 8]);           \
            v16bf b2 = frag16(&Bs[buf][32 + lr][kk + hs * 8],                 \
                              &Bs[buf][32 + lr][kk + 16 + hs * 8]);           \
            v16bf b3 = frag16(&Bs[buf][48 + lr][kk + hs * 8],                 \
                              &Bs[buf][48 + lr][kk + 16 + hs * 8]);           \
            ACC_[0] = wmma_bf16(a, b0, ACC_[0]);                              \
            ACC_[1] = wmma_bf16(a, b1, ACC_[1]);                              \
            ACC_[2] = wmma_bf16(a, b2, ACC_[2]);                              \
            ACC_[3] = wmma_bf16(a, b3, ACC_[3]);                              \
        }                                                                     \
    }

// ---------------------------------------------------------------------------
// Kernel 1: fused QKV projection GEMM + bias + layout-shuffling epilogue.
//   Q,K: [b][h][s][hd] bf16      V: [b][h][hd][s] bf16
// ---------------------------------------------------------------------------
__global__ __launch_bounds__(256) void qkv_gemm(
    const __bf16* __restrict__ A, const __bf16* __restrict__ Bt,
    const float* __restrict__ qb, const float* __restrict__ vb,
    __bf16* __restrict__ Qo, __bf16* __restrict__ Ko, __bf16* __restrict__ Vo)
{
    GEMM_CORE(A, Bt, acc)

    // C layout: lane(0-15): M=r, N=lane; lane(16-31): M=8+r, N=lane-16.
#pragma unroll
    for (int nt = 0; nt < 4; ++nt) {
        int n    = nBase + nt * 16 + lr;
        int part = n >> 10;      // 0=Q 1=K 2=V (16-col tiles never straddle)
        int rem  = n & 1023;     // h*64+hd
        int h    = rem >> 6;
        int hd   = rem & 63;
#pragma unroll
        for (int r = 0; r < 8; ++r) {
            int m = mBase + wv * 16 + r + hs * 8;
            int b = m >> 11;     // / S_
            int s = m & (S_ - 1);
            float val = acc[nt][r];
            if (part == 0) {
                val += qb[rem];
                Qo[(((size_t)(b * H_ + h)) * S_ + s) * HD_ + hd] = (__bf16)val;
            } else if (part == 1) {
                Ko[(((size_t)(b * H_ + h)) * S_ + s) * HD_ + hd] = (__bf16)val;
            } else {
                val += vb[rem];
                Vo[(((size_t)(b * H_ + h)) * HD_ + hd) * S_ + s] = (__bf16)val;
            }
        }
    }
}

// ---------------------------------------------------------------------------
// Kernel 3: output projection GEMM, fp32 out with +b_out.
// ---------------------------------------------------------------------------
__global__ __launch_bounds__(256) void out_gemm(
    const __bf16* __restrict__ A, const __bf16* __restrict__ Bt,
    const float* __restrict__ bias, float* __restrict__ out)
{
    GEMM_CORE(A, Bt, acc)

#pragma unroll
    for (int nt = 0; nt < 4; ++nt) {
        int n = nBase + nt * 16 + lr;
        float bv = bias[n];
#pragma unroll
        for (int r = 0; r < 8; ++r) {
            int m = mBase + wv * 16 + r + hs * 8;
            out[(size_t)m * D_ + n] = acc[nt][r] + bv;
        }
    }
}

// ---------------------------------------------------------------------------
// Kernel 2: flash attention. 128 threads = 4 waves; each wave owns a 16-row
// q tile. K processed in 32-column steps: 4 WMMAs (logits) + online softmax
// (half-wave shfl_xor reductions) + LDS C->A transpose of P + 4 WMMAs (P*V).
// Output X: [b][s][h*HD] bf16 (GEMM-ready for the output projection).
// ---------------------------------------------------------------------------
__global__ __launch_bounds__(128) void attn_kernel(
    const __bf16* __restrict__ Q, const __bf16* __restrict__ K,
    const __bf16* __restrict__ V, __bf16* __restrict__ X)
{
    __shared__ __bf16 plds[4][16][40];   // per-wave P bounce buffer

    const int t    = threadIdx.x;
    const int wv   = t >> 5;
    const int lane = t & 31;
    const int lr   = lane & 15;
    const int hs   = lane >> 4;
    const int qt   = blockIdx.x;
    const int h    = blockIdx.y;
    const int b    = blockIdx.z;
    const int qBase = qt * 64 + wv * 16;

    const __bf16* Qbh = Q + ((size_t)(b * H_ + h)) * S_ * HD_;
    const __bf16* Kbh = K + ((size_t)(b * H_ + h)) * S_ * HD_;
    const __bf16* Vbh = V + ((size_t)(b * H_ + h)) * HD_ * S_;

    // Q A-fragments (row = qBase+lr, d-chunks per ISA 16-bit A layout).
    const __bf16* qrow = Qbh + (size_t)(qBase + lr) * HD_;
    v16bf aq0 = frag16(qrow + hs * 8,      qrow + 16 + hs * 8);   // d 0..31
    v16bf aq1 = frag16(qrow + 32 + hs * 8, qrow + 48 + hs * 8);   // d 32..63

    float m_i[8], l_i[8];
    v8f o[4];
#pragma unroll
    for (int r = 0; r < 8; ++r) { m_i[r] = -1e30f; l_i[r] = 0.f; }
#pragma unroll
    for (int nt = 0; nt < 4; ++nt)
#pragma unroll
        for (int r = 0; r < 8; ++r) o[nt][r] = 0.f;

    for (int j = 0; j < S_; j += 32) {
        if (j + 32 < S_) {
            __builtin_prefetch(Kbh + (size_t)(j + 32 + lr) * HD_, 0, 1);
            __builtin_prefetch(Vbh + (size_t)lr * S_ + (j + 32), 0, 1);
        }

        // Logits: s0 = Q*K^T cols j..j+15, s1 = cols j+16..j+31.
        const __bf16* k0row = Kbh + (size_t)(j + lr) * HD_;
        const __bf16* k1row = Kbh + (size_t)(j + 16 + lr) * HD_;
        v16bf bk00 = frag16(k0row + hs * 8,      k0row + 16 + hs * 8);
        v16bf bk01 = frag16(k0row + 32 + hs * 8, k0row + 48 + hs * 8);
        v16bf bk10 = frag16(k1row + hs * 8,      k1row + 16 + hs * 8);
        v16bf bk11 = frag16(k1row + 32 + hs * 8, k1row + 48 + hs * 8);

        v8f s0, s1;
#pragma unroll
        for (int r = 0; r < 8; ++r) { s0[r] = 0.f; s1[r] = 0.f; }
        s0 = wmma_bf16(aq0, bk00, s0);
        s0 = wmma_bf16(aq1, bk01, s0);
        s1 = wmma_bf16(aq0, bk10, s1);
        s1 = wmma_bf16(aq1, bk11, s1);

        // Online softmax over the 32 new columns. Row M=r+8*hs lives across
        // the 16 lanes of this half-wave -> xor-reduce masks 1,2,4,8.
        float mx[8], ps[8], p0[8], p1[8], corr[8];
#pragma unroll
        for (int r = 0; r < 8; ++r) {
            s0[r] *= 0.125f;                  // 1/sqrt(HD)
            s1[r] *= 0.125f;
            mx[r] = fmaxf(s0[r], s1[r]);
        }
#pragma unroll
        for (int mask = 1; mask < 16; mask <<= 1)
#pragma unroll
            for (int r = 0; r < 8; ++r)
                mx[r] = fmaxf(mx[r], __shfl_xor(mx[r], mask, 32));
#pragma unroll
        for (int r = 0; r < 8; ++r) {
            float mn = fmaxf(m_i[r], mx[r]);
            corr[r]  = __expf(m_i[r] - mn);
            m_i[r]   = mn;
            p0[r]    = __expf(s0[r] - mn);
            p1[r]    = __expf(s1[r] - mn);
            ps[r]    = p0[r] + p1[r];
        }
#pragma unroll
        for (int mask = 1; mask < 16; mask <<= 1)
#pragma unroll
            for (int r = 0; r < 8; ++r)
                ps[r] += __shfl_xor(ps[r], mask, 32);
#pragma unroll
        for (int r = 0; r < 8; ++r) l_i[r] = l_i[r] * corr[r] + ps[r];
#pragma unroll
        for (int nt = 0; nt < 4; ++nt)
#pragma unroll
            for (int r = 0; r < 8; ++r) o[nt][r] *= corr[r];

        // C layout -> A layout transpose of P (16x32 bf16) via per-wave LDS.
#pragma unroll
        for (int r = 0; r < 8; ++r) {
            plds[wv][r + hs * 8][lr]      = (__bf16)p0[r];
            plds[wv][r + hs * 8][16 + lr] = (__bf16)p1[r];
        }
        asm volatile("s_wait_dscnt 0" ::: "memory");   // within-wave LDS RAW
        v16bf pa = frag16(&plds[wv][lr][hs * 8], &plds[wv][lr][16 + hs * 8]);

        // P(16x32) * V(32x16) per hd-tile; V stored [hd][s] so B-fragment
        // k-chunks are contiguous in s.
#pragma unroll
        for (int nt = 0; nt < 4; ++nt) {
            const __bf16* vrow = Vbh + (size_t)(nt * 16 + lr) * S_ + j;
            v16bf bv = frag16(vrow + hs * 8, vrow + 16 + hs * 8);
            o[nt] = wmma_bf16(pa, bv, o[nt]);
        }
    }

    // Normalize and write X [b][s][h*HD].
#pragma unroll
    for (int nt = 0; nt < 4; ++nt) {
#pragma unroll
        for (int r = 0; r < 8; ++r) {
            int Mrow = r + hs * 8;
            float val = o[nt][r] / l_i[r];
            X[((size_t)(b * S_ + qBase + Mrow)) * (H_ * HD_)
              + h * HD_ + nt * 16 + lr] = (__bf16)val;
        }
    }
}

// ---------------------------------------------------------------------------
// Host launcher.
// ---------------------------------------------------------------------------
extern "C" void kernel_launch(void* const* d_in, const int* in_sizes, int n_in,
                              void* d_out, int out_size, void* d_ws, size_t ws_size,
                              hipStream_t stream) {
    const float* inputs_q = (const float*)d_in[0];
    // d_in[1] = inputs_kv (unused by the reference)
    const float* W_qkv  = (const float*)d_in[2];
    const float* q_bias = (const float*)d_in[3];
    const float* v_bias = (const float*)d_in[4];
    const float* W_out  = (const float*)d_in[5];
    const float* b_out  = (const float*)d_in[6];

    char* ws = (char*)d_ws;
    size_t off = 0;
    __bf16* Abf   = (__bf16*)(ws + off); off += (size_t)M_ * D_ * 2;            // 16 MB
    __bf16* WqkvT = (__bf16*)(ws + off); off += (size_t)NQKV * D_ * 2;          //  6 MB
    __bf16* WoutT = (__bf16*)(ws + off); off += (size_t)D_ * D_ * 2;            //  2 MB
    __bf16* Qb    = (__bf16*)(ws + off); off += (size_t)B_ * H_ * S_ * HD_ * 2; // 16 MB
    __bf16* Kb    = (__bf16*)(ws + off); off += (size_t)B_ * H_ * S_ * HD_ * 2; // 16 MB
    __bf16* Vb    = (__bf16*)(ws + off); off += (size_t)B_ * H_ * S_ * HD_ * 2; // 16 MB
    __bf16* Xb    = (__bf16*)(ws + off); off += (size_t)M_ * H_ * HD_ * 2;      // 16 MB

    // 1) Precision packing (weights pre-transposed so GEMM B-fragments are
    //    contiguous along K).
    {
        int n = M_ * D_;
        cvt_f32_bf16<<<(n + 255) / 256, 256, 0, stream>>>(inputs_q, Abf, n);
    }
    {
        int n = D_ * NQKV;
        cvt_transpose_bf16<<<(n + 255) / 256, 256, 0, stream>>>(W_qkv, WqkvT, D_, NQKV);
    }
    {
        int n = D_ * D_;   // W_out is [H*HD=1024][D=1024]
        cvt_transpose_bf16<<<(n + 255) / 256, 256, 0, stream>>>(W_out, WoutT, D_, D_);
    }

    // 2) Fused QKV projection.
    qkv_gemm<<<dim3(NQKV / 64, M_ / 128), 256, 0, stream>>>(
        Abf, WqkvT, q_bias, v_bias, Qb, Kb, Vb);

    // 3) Flash attention.
    attn_kernel<<<dim3(S_ / 64, H_, B_), 128, 0, stream>>>(Qb, Kb, Vb, Xb);

    // 4) Output projection.
    out_gemm<<<dim3(D_ / 64, M_ / 128), 256, 0, stream>>>(
        Xb, WoutT, b_out, (float*)d_out);
}
